// NoResidual_60249801228690
// MI455X (gfx1250) — matrix operand order repaired and checked by reading, not compiled, not run on hardware
//
#include <hip/hip_runtime.h>
#include <hip/hip_fp16.h>

typedef __attribute__((ext_vector_type(16))) _Float16 v16h;
typedef __attribute__((ext_vector_type(8)))  float    v8f;

#define GS_LOOP(i, n) for (int i = blockIdx.x * blockDim.x + threadIdx.x; i < (n); i += gridDim.x * blockDim.x)

// ---- monotone float <-> ordered-uint mapping (for atomic float max via atomicMax(u32)) ----
__device__ __forceinline__ unsigned f2ord(float f) {
  unsigned u = __float_as_uint(f);
  return (u & 0x80000000u) ? ~u : (u | 0x80000000u);
}
__device__ __forceinline__ float ord2f(unsigned u) {
  return (u & 0x80000000u) ? __uint_as_float(u & 0x7fffffffu) : __uint_as_float(~u);
}

// ------------------------- utility fills -------------------------
__global__ void fill_f32_kernel(float* p, float v, int n) { GS_LOOP(i, n) p[i] = v; }
__global__ void fill_ord_kernel(unsigned* p, float v, int n) { unsigned o = f2ord(v); GS_LOOP(i, n) p[i] = o; }

// ------------------------- degrees -------------------------
__global__ void deg_kernel(const int* __restrict__ src, const int* __restrict__ dst,
                           float* dout, float* din, int n) {
  GS_LOOP(e, n) {
    atomicAdd(&dout[src[e]], 1.0f);
    atomicAdd(&din[dst[e]], 1.0f);
  }
}
__global__ void deg_inv_kernel(float* d, int n) {
  GS_LOOP(i, n) { float v = d[i]; d[i] = rsqrtf(v < 1.0f ? 1.0f : v); }
}

// ------------------------- fragment-order converts -------------------------
// A fragments: element ((rt*S + s)*32 + lane)*16 + j  with j = 2v+p maps to
//   A[rt*16 + (lane&15)][s*32 + (v>>2)*16 + (lane>>4)*8 + (v&3)*2 + p]
// grid.x = row_tiles * S, block = 512 (lane = tid>>4, j = tid&15)
template <int K>
__global__ void __launch_bounds__(512) cvt_a_frag_kernel(
    const float* __restrict__ in, const float* __restrict__ scale,
    _Float16* __restrict__ out, int n) {
  constexpr int S = K / 32;
  const int tid = threadIdx.x;
  const int lane = tid >> 4, j = tid & 15;
  const int rt = blockIdx.x / S, s = blockIdx.x - rt * S;
  const int m = lane & 15, half = lane >> 4;
  const int v = j >> 1, p = j & 1;
  const int row = rt * 16 + m;
  const int k = s * 32 + (v >> 2) * 16 + half * 8 + (v & 3) * 2 + p;
  float val = 0.0f;
  if (row < n) val = in[(size_t)row * K + k] * (scale ? scale[row] : 1.0f);
  out[(size_t)blockIdx.x * 512 + tid] = (_Float16)val;
}

// W fragments: element ((s*6 + ct)*32 + lane)*16 + j  with j = 2v+p maps to
//   W[s*32 + (v>>2)*16 + (lane>>4)*8 + (v&3)*2 + p][ct*16 + (lane&15)]
// grid.x = S * 6, block = 512
template <int K>
__global__ void __launch_bounds__(512) cvt_w_frag_kernel(
    const float* __restrict__ W, _Float16* __restrict__ out) {
  const int tid = threadIdx.x;
  const int lane = tid >> 4, j = tid & 15;
  const int s = blockIdx.x / 6, ct = blockIdx.x - s * 6;
  const int m = lane & 15, half = lane >> 4;
  const int v = j >> 1, p = j & 1;
  const int k = s * 32 + (v >> 2) * 16 + half * 8 + (v & 3) * 2 + p;
  const int col = ct * 16 + m;
  out[(size_t)blockIdx.x * 512 + tid] = (_Float16)W[(size_t)k * 96 + col];
}

// ------------------------- WMMA GEMM over fragment buffers -------------------------
// 8 waves/block, one 16-row strip per wave; 6 column tiles held in accumulators.
template <int K>
__global__ void __launch_bounds__(256) wmma_gemm96_kernel(
    const v16h* __restrict__ Af, const v16h* __restrict__ Bf,
    const float* __restrict__ bias, float* __restrict__ out, int row_tiles) {
  constexpr int S = K / 32;
  const int tid = threadIdx.x;
  const int lane = tid & 31, wave = tid >> 5;
  const int rt = blockIdx.x * 8 + wave;
  if (rt >= row_tiles) return;
  const v16h* ap = Af + (size_t)rt * S * 32 + lane;
  const v16h* bp = Bf + lane;
  v8f acc[6] = {};
#pragma unroll
  for (int s = 0; s < S; ++s) {
    v16h a = ap[(size_t)s * 32];
#pragma unroll
    for (int ct = 0; ct < 6; ++ct) {
      v16h b = bp[(size_t)(s * 6 + ct) * 32];
      acc[ct] = __builtin_amdgcn_wmma_f32_16x16x32_f16(false, a, false, b, (short)0, acc[ct], false, false);
    }
  }
  // C/D layout: lane L -> col = L&15; VGPR r -> row = 8*(L>>4) + r
  const int m = lane & 15, half = lane >> 4;
  float* orow = out + ((size_t)rt * 16 + (size_t)half * 8) * 96 + m;
#pragma unroll
  for (int ct = 0; ct < 6; ++ct) {
    const float bv = bias ? bias[ct * 16 + m] : 0.0f;
#pragma unroll
    for (int r = 0; r < 8; ++r) orow[(size_t)r * 96 + ct * 16] = acc[ct][r] + bv;
  }
}

// ------------------------- GraphConv edge aggregation -------------------------
__global__ void edge_agg_kernel(const float* __restrict__ t, const int* __restrict__ src,
                                const int* __restrict__ dst, float* agg, int total) {
  GS_LOOP(idx, total) {
    int e = idx / 96, k = idx - e * 96;
    atomicAdd(&agg[(size_t)dst[e] * 96 + k], t[(size_t)src[e] * 96 + k]);
  }
}
__global__ void gc_final_kernel(const float* __restrict__ agg, const float* __restrict__ dinv,
                                const float* __restrict__ b, float* h, int total) {
  GS_LOOP(idx, total) {
    int i = idx / 96, k = idx - i * 96;
    float v = agg[idx] * dinv[i] + b[k];
    h[idx] = v > 0.0f ? v : 0.0f;
  }
}

// ------------------------- GATv2 edge kernels -------------------------
__global__ void gat_logits_kernel(const float* __restrict__ fs, const float* __restrict__ fd,
                                  const float* __restrict__ attn, const int* __restrict__ src,
                                  const int* __restrict__ dst, float* __restrict__ logits,
                                  unsigned* mord, int total) {
  GS_LOOP(idx, total) {
    int e = idx >> 3, hh = idx & 7;
    const float* ps = fs + (size_t)src[e] * 96 + hh * 12;
    const float* pd = fd + (size_t)dst[e] * 96 + hh * 12;
    const float* pa = attn + hh * 12;
    float s = 0.0f;
#pragma unroll
    for (int d0 = 0; d0 < 12; ++d0) {
      float t = ps[d0] + pd[d0];
      t = t > 0.0f ? t : 0.2f * t;  // leaky_relu(0.2)
      s += pa[d0] * t;
    }
    logits[idx] = s;
    atomicMax(&mord[(size_t)dst[e] * 8 + hh], f2ord(s));
  }
}
__global__ void gat_exp_kernel(const float* __restrict__ logits, const unsigned* __restrict__ mord,
                               const int* __restrict__ dst, float* ex, float* den, int total) {
  GS_LOOP(idx, total) {
    int e = idx >> 3, hh = idx & 7;
    int d = dst[e];
    float v = expf(logits[idx] - ord2f(mord[(size_t)d * 8 + hh]));
    ex[idx] = v;
    atomicAdd(&den[(size_t)d * 8 + hh], v);
  }
}
__global__ void gat_agg_kernel(const float* __restrict__ ex, const float* __restrict__ den,
                               const float* __restrict__ fs, const int* __restrict__ src,
                               const int* __restrict__ dst, float* rst, int total) {
  GS_LOOP(idx, total) {
    int e = idx / 96, k = idx - e * 96;
    int hh = k / 12;
    int d = dst[e];
    float a = ex[(size_t)e * 8 + hh] / den[(size_t)d * 8 + hh];
    atomicAdd(&rst[(size_t)d * 96 + k], a * fs[(size_t)src[e] * 96 + k]);
  }
}
__global__ void relu_kernel(const float* __restrict__ in, float* out, int n) {
  GS_LOOP(i, n) { float v = in[i]; out[i] = v > 0.0f ? v : 0.0f; }
}

// ------------------------- readout + MLP -------------------------
__global__ void readout_kernel(const float* __restrict__ h, const int* __restrict__ gid,
                               unsigned* hgord, int total) {
  GS_LOOP(idx, total) {
    int i = idx / 96, k = idx - i * 96;
    atomicMax(&hgord[(size_t)gid[i] * 96 + k], f2ord(h[idx]));
  }
}
__global__ void mlp1_kernel(const unsigned* __restrict__ hgord, const float* __restrict__ W,
                            const float* __restrict__ b, float* z) {
  GS_LOOP(idx, 64 * 96) {
    int g = idx / 96, j = idx - g * 96;
    float s = b[j];
    for (int k = 0; k < 96; ++k) s += ord2f(hgord[g * 96 + k]) * W[k * 96 + j];
    z[idx] = s > 0.0f ? s : 0.0f;
  }
}
__global__ void mlp2_kernel(const float* __restrict__ z1, const float* __restrict__ W,
                            const float* __restrict__ b, float* z) {
  GS_LOOP(idx, 64 * 48) {
    int g = idx / 48, j = idx - g * 48;
    float s = b[j];
    for (int k = 0; k < 96; ++k) s += z1[g * 96 + k] * W[k * 48 + j];
    z[idx] = s > 0.0f ? s : 0.0f;
  }
}
__global__ void mlp3_kernel(const float* __restrict__ z2, const float* __restrict__ W,
                            const float* __restrict__ b, float* out) {
  GS_LOOP(idx, 64 * 10) {
    int g = idx / 10, j = idx - g * 10;
    float s = b[j];
    for (int k = 0; k < 48; ++k) s += z2[g * 48 + k] * W[k * 10 + j];
    out[idx] = s;
  }
}

// ------------------------- host orchestration -------------------------
extern "C" void kernel_launch(void* const* d_in, const int* in_sizes, int n_in,
                              void* d_out, int out_size, void* d_ws, size_t ws_size,
                              hipStream_t stream) {
  constexpr int N = 50000, E = 800000, IN = 128, HID = 96, H = 8, G = 64;
  constexpr int NPAD = 50048;       // multiple of 64
  constexpr int RT = NPAD / 16;     // 3128 row tiles

  const float* x    = (const float*)d_in[0];
  const int*   src  = (const int*)d_in[1];
  const int*   dst  = (const int*)d_in[2];
  const int*   gid  = (const int*)d_in[3];
  const float* W1   = (const float*)d_in[4];
  const float* b1   = (const float*)d_in[5];
  const float* W2   = (const float*)d_in[6];
  const float* b2   = (const float*)d_in[7];
  const float* Ws   = (const float*)d_in[8];
  const float* bs   = (const float*)d_in[9];
  const float* Wd   = (const float*)d_in[10];
  const float* bd   = (const float*)d_in[11];
  const float* attn = (const float*)d_in[12];
  const float* Wc1  = (const float*)d_in[13];
  const float* bc1  = (const float*)d_in[14];
  const float* Wc2  = (const float*)d_in[15];
  const float* bc2  = (const float*)d_in[16];
  const float* Wc3  = (const float*)d_in[17];
  const float* bc3  = (const float*)d_in[18];
  float* out = (float*)d_out;

  // bump allocator over workspace
  char* wp = (char*)d_ws;
  auto alloc = [&](size_t bytes) -> void* {
    void* r = (void*)wp;
    wp += (bytes + 255) & ~(size_t)255;
    return r;
  };
  float*     deg_out = (float*)alloc((size_t)N * 4);
  float*     deg_in  = (float*)alloc((size_t)N * 4);
  _Float16*  A_h     = (_Float16*)alloc((size_t)NPAD * 128 * 2);
  _Float16*  W_h     = (_Float16*)alloc((size_t)128 * 96 * 2);
  float*     tmp     = (float*)alloc((size_t)NPAD * 96 * 4);
  float*     agg     = (float*)alloc((size_t)NPAD * 96 * 4);
  float*     hbuf    = (float*)alloc((size_t)NPAD * 96 * 4);
  float*     fsb     = (float*)alloc((size_t)NPAD * 96 * 4);
  float*     fdb     = (float*)alloc((size_t)NPAD * 96 * 4);
  float*     logits  = (float*)alloc((size_t)E * 8 * 4);
  float*     exb     = (float*)alloc((size_t)E * 8 * 4);
  unsigned*  mord    = (unsigned*)alloc((size_t)N * 8 * 4);
  float*     den     = (float*)alloc((size_t)N * 8 * 4);
  unsigned*  hgord   = (unsigned*)alloc((size_t)G * 96 * 4);
  float*     z1      = (float*)alloc((size_t)G * 96 * 4);
  float*     z2      = (float*)alloc((size_t)G * 48 * 4);

  const int TB = 256;
  auto blocks = [](int n, int tb) { return (n + tb - 1) / tb; };

  // ---- degrees ----
  fill_f32_kernel<<<blocks(N, TB), TB, 0, stream>>>(deg_out, 0.0f, N);
  fill_f32_kernel<<<blocks(N, TB), TB, 0, stream>>>(deg_in, 0.0f, N);
  deg_kernel<<<blocks(E, TB), TB, 0, stream>>>(src, dst, deg_out, deg_in, E);
  deg_inv_kernel<<<blocks(N, TB), TB, 0, stream>>>(deg_out, N);
  deg_inv_kernel<<<blocks(N, TB), TB, 0, stream>>>(deg_in, N);

  const int gemm_blocks = (RT + 7) / 8;
  const int n96 = NPAD * 96;
  const int e96 = E * 96;
  const int e8  = E * 8;

  // ---- GraphConv 1 (K = 128, S = 4) ----
  cvt_a_frag_kernel<128><<<RT * 4, 512, 0, stream>>>(x, deg_out, A_h, N);
  cvt_w_frag_kernel<128><<<4 * 6, 512, 0, stream>>>(W1, W_h);
  wmma_gemm96_kernel<128><<<gemm_blocks, 256, 0, stream>>>((const v16h*)A_h, (const v16h*)W_h, nullptr, tmp, RT);
  fill_f32_kernel<<<blocks(n96, TB), TB, 0, stream>>>(agg, 0.0f, n96);
  edge_agg_kernel<<<blocks(e96, TB), TB, 0, stream>>>(tmp, src, dst, agg, e96);
  gc_final_kernel<<<blocks(N * 96, TB), TB, 0, stream>>>(agg, deg_in, b1, hbuf, N * 96);

  // ---- GraphConv 2 (K = 96, S = 3) ----
  cvt_a_frag_kernel<96><<<RT * 3, 512, 0, stream>>>(hbuf, deg_out, A_h, N);
  cvt_w_frag_kernel<96><<<3 * 6, 512, 0, stream>>>(W2, W_h);
  wmma_gemm96_kernel<96><<<gemm_blocks, 256, 0, stream>>>((const v16h*)A_h, (const v16h*)W_h, nullptr, tmp, RT);
  fill_f32_kernel<<<blocks(n96, TB), TB, 0, stream>>>(agg, 0.0f, n96);
  edge_agg_kernel<<<blocks(e96, TB), TB, 0, stream>>>(tmp, src, dst, agg, e96);
  gc_final_kernel<<<blocks(N * 96, TB), TB, 0, stream>>>(agg, deg_in, b2, hbuf, N * 96);

  // ---- 3x GATv2 ----
  for (int layer = 0; layer < 3; ++layer) {
    const float* Wsl = Ws + (size_t)layer * HID * HID;
    const float* bsl = bs + (size_t)layer * HID;
    const float* Wdl = Wd + (size_t)layer * HID * HID;
    const float* bdl = bd + (size_t)layer * HID;
    const float* al  = attn + (size_t)layer * H * 12;

    cvt_a_frag_kernel<96><<<RT * 3, 512, 0, stream>>>(hbuf, nullptr, A_h, N);
    cvt_w_frag_kernel<96><<<3 * 6, 512, 0, stream>>>(Wsl, W_h);
    wmma_gemm96_kernel<96><<<gemm_blocks, 256, 0, stream>>>((const v16h*)A_h, (const v16h*)W_h, bsl, fsb, RT);
    cvt_w_frag_kernel<96><<<3 * 6, 512, 0, stream>>>(Wdl, W_h);
    wmma_gemm96_kernel<96><<<gemm_blocks, 256, 0, stream>>>((const v16h*)A_h, (const v16h*)W_h, bdl, fdb, RT);

    fill_ord_kernel<<<blocks(N * 8, TB), TB, 0, stream>>>(mord, -3.0e38f, N * 8);
    gat_logits_kernel<<<blocks(e8, TB), TB, 0, stream>>>(fsb, fdb, al, src, dst, logits, mord, e8);
    fill_f32_kernel<<<blocks(N * 8, TB), TB, 0, stream>>>(den, 0.0f, N * 8);
    gat_exp_kernel<<<blocks(e8, TB), TB, 0, stream>>>(logits, mord, dst, exb, den, e8);
    fill_f32_kernel<<<blocks(n96, TB), TB, 0, stream>>>(tmp, 0.0f, n96);
    gat_agg_kernel<<<blocks(e96, TB), TB, 0, stream>>>(exb, den, fsb, src, dst, tmp, e96);
    relu_kernel<<<blocks(N * 96, TB), TB, 0, stream>>>(tmp, hbuf, N * 96);
  }

  // ---- per-graph max readout + MLP ----
  fill_ord_kernel<<<blocks(G * 96, TB), TB, 0, stream>>>(hgord, -3.0e38f, G * 96);
  readout_kernel<<<blocks(N * 96, TB), TB, 0, stream>>>(hbuf, gid, hgord, N * 96);
  mlp1_kernel<<<blocks(G * 96, TB), TB, 0, stream>>>(hgord, Wc1, bc1, z1);
  mlp2_kernel<<<blocks(G * 48, TB), TB, 0, stream>>>(z1, Wc2, bc2, z2);
  mlp3_kernel<<<blocks(G * 10, TB), TB, 0, stream>>>(z2, Wc3, bc3, out);
}
